// AgentAwareAttentionV2_29746943492472
// MI455X (gfx1250) — compile-verified, matrix operand order checked by reading, not run on hardware
//
#include <hip/hip_runtime.h>

// ---------------------------------------------------------------------------
// AgentAwareAttention for MI455X (gfx1250), wave32 + v_wmma_f32_16x16x32_f16.
// f16 data / f32 accumulation everywhere; softmax math in f32.
// ---------------------------------------------------------------------------

typedef __attribute__((ext_vector_type(16))) _Float16 v16h;
typedef __attribute__((ext_vector_type(8)))  float    v8f;
typedef __attribute__((ext_vector_type(2)))  _Float16 h2;

#define L_SEQ 2048
#define S_SEQ 2048
#define NBATCH 2
#define DMODEL 512
#define NHEAD 8
#define HDIM 64

// ---- WMMA helpers ----------------------------------------------------------

__device__ __forceinline__ v8f wmma_f16(v16h a, v16h b, v8f c) {
  // D = A(16x32 f16) * B(32x16 f16) + C(16x16 f32)
  return __builtin_amdgcn_wmma_f32_16x16x32_f16(
      /*neg_a=*/false, a, /*neg_b=*/false, b,
      /*c_mod=*/(short)0, c, /*reuse_a=*/false, /*reuse_b=*/false);
}

// A-frag: lane holds row (lane&15); 16 K-elements with pattern
//   kb = (j>>2)*16 + (lane>>4)*8 + (j&3)*2   (pairs kb, kb+1)
// B-frag: identical pattern with lane&15 = output column, reading the
// column-major row (i.e. B_col[n][k]).
__device__ __forceinline__ v16h frag_from_f16row(const _Float16* __restrict__ rowPtr,
                                                 int khalf) {
  v16h f;
#pragma unroll
  for (int j = 0; j < 8; ++j) {
    int kb = ((j >> 2) << 4) + (khalf << 3) + ((j & 3) << 1);
    h2 p = *(const h2*)(rowPtr + kb);      // 4-byte load (kb is even)
    f[2 * j]     = p.x;
    f[2 * j + 1] = p.y;
  }
  return f;
}

__device__ __forceinline__ v16h frag_from_f32row(const float* __restrict__ rowPtr,
                                                 int khalf, float scale) {
  v16h f;
#pragma unroll
  for (int j = 0; j < 8; ++j) {
    int kb = ((j >> 2) << 4) + (khalf << 3) + ((j & 3) << 1);
    f[2 * j]     = (_Float16)(rowPtr[kb] * scale);
    f[2 * j + 1] = (_Float16)(rowPtr[kb + 1] * scale);
  }
  return f;
}

// ---------------------------------------------------------------------------
// Kernel 1: the five input projections  Y = X @ W^T  (4096 x 512 x 512 each),
// fp32 in, f16 out, scattered into attention-friendly layouts.
//   p=0: Qs[h][n][l][hd]  (x 1/sqrt(HD))
//   p=1: Qo[h][n][l][hd]  (x 1/sqrt(HD))
//   p=2: Ks[h][n][s][hd]
//   p=3: Ko[h][n][s][hd]
//   p=4: Vt[h][n][hd][s]   (transposed for AV B-fragments)
// grid: (4096/128, 512/64, 5), 256 threads (8 waves, 16x64 tile per wave)
// ---------------------------------------------------------------------------
__global__ __launch_bounds__(256) void proj_kernel(
    const float* __restrict__ q, const float* __restrict__ k,
    const float* __restrict__ v,
    const float* __restrict__ Wqs, const float* __restrict__ Wqo,
    const float* __restrict__ Wks, const float* __restrict__ Wko,
    const float* __restrict__ Wv,
    _Float16* __restrict__ Qs, _Float16* __restrict__ Qo,
    _Float16* __restrict__ Ks, _Float16* __restrict__ Ko,
    _Float16* __restrict__ Vt) {
  const int lane  = threadIdx.x & 31;
  const int wid   = threadIdx.x >> 5;
  const int col   = lane & 15;
  const int khalf = lane >> 4;
  const int p     = blockIdx.z;
  const int m0    = blockIdx.x * 128 + wid * 16;   // row = t*NBATCH + n
  const int e0    = blockIdx.y * 64;               // output feature tile

  const float* X = (p < 2) ? q : ((p < 4) ? k : v);
  const float* W = (p == 0) ? Wqs : (p == 1) ? Wqo : (p == 2) ? Wks
                 : (p == 3) ? Wko : Wv;
  _Float16* dst  = (p == 0) ? Qs : (p == 1) ? Qo : (p == 2) ? Ks
                 : (p == 3) ? Ko : Vt;
  const float scale = (p < 2) ? 0.125f : 1.0f;     // HD^-0.5 folded into Q

  const float* Xrow = X + (size_t)(m0 + col) * DMODEL;
  const float* Wr[4];
#pragma unroll
  for (int et = 0; et < 4; ++et)
    Wr[et] = W + (size_t)(e0 + et * 16 + col) * DMODEL;

  v8f cc[4] = {v8f{}, v8f{}, v8f{}, v8f{}};
  for (int k0 = 0; k0 < DMODEL; k0 += 32) {
    v16h a = frag_from_f32row(Xrow + k0, khalf, scale);
#pragma unroll
    for (int et = 0; et < 4; ++et) {
      v16h b = frag_from_f32row(Wr[et] + k0, khalf, 1.0f);
      cc[et] = wmma_f16(a, b, cc[et]);
    }
  }

#pragma unroll
  for (int et = 0; et < 4; ++et) {
#pragma unroll
    for (int vr = 0; vr < 8; ++vr) {
      int m   = vr + khalf * 8;          // C layout: row = vgpr + half*8
      int rid = m0 + m;
      int t   = rid >> 1;                // rows are (t,n) flattened, N=2
      int nb  = rid & 1;
      int e   = e0 + et * 16 + col;
      int h   = e >> 6;
      int hd  = e & 63;
      _Float16 val = (_Float16)cc[et][vr];
      if (p != 4)
        dst[(((size_t)(h * NBATCH + nb) * 2048 + t) << 6) + hd] = val;
      else
        dst[((size_t)(h * NBATCH + nb) * 64 + hd) * 2048 + t] = val;
    }
  }
}

// ---------------------------------------------------------------------------
// Logit tile helper: 16(L) x 16(S) agent-aware logits in C layout.
// ---------------------------------------------------------------------------
__device__ __forceinline__ void logits_tile(
    const _Float16* __restrict__ Ksb, const _Float16* __restrict__ Kob,
    const int* __restrict__ kid_g, const float* __restrict__ mask,
    v16h qsA0, v16h qsA1, v16h qoA0, v16h qoA1,
    const int* qid, int l0, int s0, int col, int khalf, float* x) {
  v16h kB0 = frag_from_f16row(Ksb + (size_t)(s0 + col) * HDIM, khalf);
  v16h kB1 = frag_from_f16row(Ksb + (size_t)(s0 + col) * HDIM + 32, khalf);
  v8f cS = {};
  cS = wmma_f16(qsA0, kB0, cS);
  cS = wmma_f16(qsA1, kB1, cS);
  v16h oB0 = frag_from_f16row(Kob + (size_t)(s0 + col) * HDIM, khalf);
  v16h oB1 = frag_from_f16row(Kob + (size_t)(s0 + col) * HDIM + 32, khalf);
  v8f cO = {};
  cO = wmma_f16(qoA0, oB0, cO);
  cO = wmma_f16(qoA1, oB1, cO);
  int kid = kid_g[s0 + col];
#pragma unroll
  for (int vr = 0; vr < 8; ++vr) {
    int row  = vr + khalf * 8;
    float mv = mask[(size_t)(l0 + row) * S_SEQ + s0 + col];
    x[vr] = ((kid == qid[vr]) ? cS[vr] : cO[vr]) + mv;
  }
}

// ---------------------------------------------------------------------------
// Kernel 2: attention. grid (L/32, N), 64 threads = 2 waves, each wave owns a
// 16-row L tile. Heads looped INSIDE the block -> deterministic in-order RMW
// accumulation of att_mean (no atomics), O written per-head 64-col slice.
// Two passes over S per head: pass1 online max/sum stats; pass2 recompute,
// emit probabilities and accumulate O = P @ V via WMMA (P staged via LDS to
// convert C-fragment layout -> A-fragment layout).
// ---------------------------------------------------------------------------
__global__ __launch_bounds__(64) void attn_kernel(
    const _Float16* __restrict__ Qs, const _Float16* __restrict__ Qo,
    const _Float16* __restrict__ Ks, const _Float16* __restrict__ Ko,
    const _Float16* __restrict__ Vt,
    const int* __restrict__ qid_g, const int* __restrict__ kid_g,
    const float* __restrict__ mask,
    _Float16* __restrict__ Obuf, float* __restrict__ att_mean) {
  const int lane  = threadIdx.x & 31;
  const int wid   = threadIdx.x >> 5;
  const int col   = lane & 15;
  const int khalf = lane >> 4;
  const int n     = blockIdx.y;
  const int l0    = blockIdx.x * 32 + wid * 16;

  // Per-wave P staging: 16 x 32 f16, row stride 34 elems (68B) so that the
  // A-fragment column reads hit distinct LDS banks (17*r mod 64 distinct).
  __shared__ __attribute__((aligned(16))) _Float16 Pb[2][16 * 34];
  _Float16* myP = &Pb[wid][0];

  int qid[8];
#pragma unroll
  for (int vr = 0; vr < 8; ++vr) qid[vr] = qid_g[l0 + vr + khalf * 8];

  for (int h = 0; h < NHEAD; ++h) {
    const _Float16* Qsb = Qs + ((size_t)(h * NBATCH + n) * L_SEQ + l0) * HDIM;
    const _Float16* Qob = Qo + ((size_t)(h * NBATCH + n) * L_SEQ + l0) * HDIM;
    const _Float16* Ksb = Ks + (size_t)(h * NBATCH + n) * S_SEQ * HDIM;
    const _Float16* Kob = Ko + (size_t)(h * NBATCH + n) * S_SEQ * HDIM;
    const _Float16* Vtb = Vt + (size_t)(h * NBATCH + n) * HDIM * S_SEQ;

    v16h qsA0 = frag_from_f16row(Qsb + col * HDIM, khalf);
    v16h qsA1 = frag_from_f16row(Qsb + col * HDIM + 32, khalf);
    v16h qoA0 = frag_from_f16row(Qob + col * HDIM, khalf);
    v16h qoA1 = frag_from_f16row(Qob + col * HDIM + 32, khalf);

    // ---- pass 1: online softmax statistics over S -------------------------
    float mrow[8], srow[8];
#pragma unroll
    for (int vr = 0; vr < 8; ++vr) { mrow[vr] = -1e30f; srow[vr] = 0.0f; }

    for (int st = 0; st < S_SEQ / 16; ++st) {
      int s0 = st * 16;
      float x[8];
      logits_tile(Ksb, Kob, kid_g, mask, qsA0, qsA1, qoA0, qoA1,
                  qid, l0, s0, col, khalf, x);
      float red[8];
#pragma unroll
      for (int vr = 0; vr < 8; ++vr) red[vr] = x[vr];
#pragma unroll
      for (int d = 1; d < 16; d <<= 1)
#pragma unroll
        for (int vr = 0; vr < 8; ++vr)
          red[vr] = fmaxf(red[vr], __shfl_xor(red[vr], d, 32));
      float es[8];
#pragma unroll
      for (int vr = 0; vr < 8; ++vr) {
        float mn = fmaxf(mrow[vr], red[vr]);
        es[vr]   = __expf(x[vr] - mn);
        red[vr]  = mn;                       // reuse: new running max
      }
#pragma unroll
      for (int d = 1; d < 16; d <<= 1)
#pragma unroll
        for (int vr = 0; vr < 8; ++vr) es[vr] += __shfl_xor(es[vr], d, 32);
#pragma unroll
      for (int vr = 0; vr < 8; ++vr) {
        srow[vr] = srow[vr] * __expf(mrow[vr] - red[vr]) + es[vr];
        mrow[vr] = red[vr];
      }
    }
    float invl[8];
#pragma unroll
    for (int vr = 0; vr < 8; ++vr) invl[vr] = 1.0f / srow[vr];

    // ---- pass 2: probabilities -> att_mean, and O += P @ V ---------------
    v8f oc[4] = {v8f{}, v8f{}, v8f{}, v8f{}};

    for (int sc = 0; sc < S_SEQ / 32; ++sc) {
#pragma unroll
      for (int sub = 0; sub < 2; ++sub) {
        int s0 = sc * 32 + sub * 16;
        float x[8];
        logits_tile(Ksb, Kob, kid_g, mask, qsA0, qsA1, qoA0, qoA1,
                    qid, l0, s0, col, khalf, x);
#pragma unroll
        for (int vr = 0; vr < 8; ++vr) {
          int row = vr + khalf * 8;
          float pp = __expf(x[vr] - mrow[vr]) * invl[vr];
          myP[row * 34 + sub * 16 + col] = (_Float16)pp;
          size_t ai = ((size_t)n * L_SEQ + (l0 + row)) * (size_t)S_SEQ
                      + s0 + col;
          float contrib = pp * (1.0f / NHEAD);
          att_mean[ai] = (h == 0) ? contrib : (att_mean[ai] + contrib);
        }
      }
      // A-fragment of P (16 x 32) from LDS (same wave: DS ops are in-order).
      v16h aP;
      {
        const _Float16* rp = myP + col * 34;
#pragma unroll
        for (int j = 0; j < 8; ++j) {
          int kb = ((j >> 2) << 4) + (khalf << 3) + ((j & 3) << 1);
          aP[2 * j]     = rp[kb];
          aP[2 * j + 1] = rp[kb + 1];
        }
      }
#pragma unroll
      for (int dt = 0; dt < 4; ++dt) {
        const _Float16* vrow = Vtb + (size_t)(dt * 16 + col) * S_SEQ + sc * 32;
        v16h bV = frag_from_f16row(vrow, khalf);
        oc[dt] = wmma_f16(aP, bV, oc[dt]);
      }
    }

    // store this head's 64-column slice of O (rows l0..l0+15)
#pragma unroll
    for (int dt = 0; dt < 4; ++dt)
#pragma unroll
      for (int vr = 0; vr < 8; ++vr) {
        int row    = vr + khalf * 8;
        size_t rid = (size_t)(l0 + row) * NBATCH + n;
        int e      = h * HDIM + dt * 16 + col;
        Obuf[rid * DMODEL + e] = (_Float16)oc[dt][vr];
      }
  }
}

// ---------------------------------------------------------------------------
// Kernel 3: final FC  out = O @ Wfc^T  (4096 x 512 x 512), f16 A, f32 out.
// grid (32, 8), 256 threads (8 waves, 16x64 tile per wave).
// ---------------------------------------------------------------------------
__global__ __launch_bounds__(256) void fc_kernel(
    const _Float16* __restrict__ Obuf, const float* __restrict__ Wfc,
    float* __restrict__ out) {
  const int lane  = threadIdx.x & 31;
  const int wid   = threadIdx.x >> 5;
  const int col   = lane & 15;
  const int khalf = lane >> 4;
  const int m0    = blockIdx.x * 128 + wid * 16;
  const int e0    = blockIdx.y * 64;

  const _Float16* Arow = Obuf + (size_t)(m0 + col) * DMODEL;
  const float* Wr[4];
#pragma unroll
  for (int et = 0; et < 4; ++et)
    Wr[et] = Wfc + (size_t)(e0 + et * 16 + col) * DMODEL;

  v8f cc[4] = {v8f{}, v8f{}, v8f{}, v8f{}};
  for (int k0 = 0; k0 < DMODEL; k0 += 32) {
    v16h a = frag_from_f16row(Arow + k0, khalf);
#pragma unroll
    for (int et = 0; et < 4; ++et) {
      v16h b = frag_from_f32row(Wr[et] + k0, khalf, 1.0f);
      cc[et] = wmma_f16(a, b, cc[et]);
    }
  }
#pragma unroll
  for (int et = 0; et < 4; ++et)
#pragma unroll
    for (int vr = 0; vr < 8; ++vr) {
      int m   = vr + khalf * 8;
      int rid = m0 + m;
      int e   = e0 + et * 16 + col;
      out[(size_t)rid * DMODEL + e] = cc[et][vr];
    }
}

// ---------------------------------------------------------------------------
// Host launcher
// ---------------------------------------------------------------------------
extern "C" void kernel_launch(void* const* d_in, const int* in_sizes, int n_in,
                              void* d_out, int out_size, void* d_ws,
                              size_t ws_size, hipStream_t stream) {
  (void)in_sizes; (void)n_in; (void)out_size; (void)ws_size;
  const float* q    = (const float*)d_in[0];
  const float* k    = (const float*)d_in[1];
  const float* v    = (const float*)d_in[2];
  const int*   qid  = (const int*)d_in[3];
  const int*   kid  = (const int*)d_in[4];
  const float* mask = (const float*)d_in[5];
  const float* Wqs  = (const float*)d_in[6];
  const float* Wqo  = (const float*)d_in[7];
  const float* Wks  = (const float*)d_in[8];
  const float* Wko  = (const float*)d_in[9];
  const float* Wv   = (const float*)d_in[10];
  const float* Wfc  = (const float*)d_in[11];

  const size_t BUF = (size_t)NHEAD * NBATCH * 2048 * HDIM;  // 2,097,152 f16
  _Float16* ws = (_Float16*)d_ws;
  _Float16* Qs = ws;
  _Float16* Qo = Qs + BUF;
  _Float16* Ks = Qo + BUF;
  _Float16* Ko = Ks + BUF;
  _Float16* Vt = Ko + BUF;
  _Float16* Ob = Vt + BUF;          // 24 MB total workspace

  float* out      = (float*)d_out;                               // [L,N,D]
  float* att_mean = out + (size_t)L_SEQ * NBATCH * DMODEL;       // [N,L,S]

  proj_kernel<<<dim3(32, 8, 5), 256, 0, stream>>>(
      q, k, v, Wqs, Wqo, Wks, Wko, Wv, Qs, Qo, Ks, Ko, Vt);
  attn_kernel<<<dim3(L_SEQ / 32, NBATCH), 64, 0, stream>>>(
      Qs, Qo, Ks, Ko, Vt, qid, kid, mask, Ob, att_mean);
  fc_kernel<<<dim3(32, 8), 256, 0, stream>>>(Ob, Wfc, out);
}
